// Graph2Feat_53068615910293
// MI455X (gfx1250) — matrix operand order repaired and checked by reading, not compiled, not run on hardware
//
#include <hip/hip_runtime.h>
#include <hip/hip_bf16.h>

// Hetero-GraphSAGE inference for MI455X (gfx1250, wave32, WMMA).
//   z_b = 0.5*(mean_ab@Wl_ab + mean_cb@Wl_cb + x_b@(Wr_ab+Wr_cb) + bl_ab + bl_cb)
//   z_c =      mean_ac@Wl_ac + x_c@Wr_ac + bl_ac
// Aggregation: f32 hardware atomics (memory-bound, ~45us floor at 23.3 TB/s).
// GEMMs: V_WMMA_F32_16X16X32_F16 on the XDL pipe with a 2-term f16 split
// (x = hi+lo; x@W ~= hi@Whi + hi@Wlo + lo@Whi, ~1e-6 relative error).
// A-rows are split to f16 hi/lo ONCE per block into LDS (shared by all 8
// waves), killing the 8x redundant conversion VALU of the naive version.

#define DFEAT 128
#define PSTR  392   // halves per LDS plane: 16 entries * 24 (16 used + 8 pad) + 8

typedef _Float16 v16h __attribute__((ext_vector_type(16)));
typedef _Float16 v8h  __attribute__((ext_vector_type(8)));
typedef float    v8f  __attribute__((ext_vector_type(8)));

// ---------------------------------------------------------------------------
// Edge aggregation: one wave (32 lanes) per edge; each lane moves float4.
// acc[dst] += x_src[src]; cnt[dst] += 1
// ---------------------------------------------------------------------------
__global__ void __launch_bounds__(256) agg_edges(const float* __restrict__ xsrc,
                                                 const int* __restrict__ src,
                                                 const int* __restrict__ dst,
                                                 float* __restrict__ acc,
                                                 float* __restrict__ cnt,
                                                 int nE) {
  const int e    = blockIdx.x * 8 + (threadIdx.x >> 5);
  const int lane = threadIdx.x & 31;
  if (e >= nE) return;
  const int s = src[e];
  const int d = dst[e];
  const float4 v = *(reinterpret_cast<const float4*>(xsrc + (size_t)s * DFEAT) + lane);
  float* a = acc + (size_t)d * DFEAT + lane * 4;
  unsafeAtomicAdd(a + 0, v.x);
  unsafeAtomicAdd(a + 1, v.y);
  unsafeAtomicAdd(a + 2, v.z);
  unsafeAtomicAdd(a + 3, v.w);
  if (lane == 0) unsafeAtomicAdd(cnt + d, 1.0f);
}

// ---------------------------------------------------------------------------
// Prep: fuse lin_r matrices/biases for the b-target relations and emit each
// weight matrix as hi/lo f16 planes pre-swizzled to this code's B-fragment order for
// V_WMMA_F32_16X16X32_F16 (contiguous k, same order as the LDS A side): for K-block kb, lane-half h,
// column n, fragment element e holds W[kb*32+16h+e][n].
// Packed index: (((kb*2+h)*128)+n)*16 + e  -> one 32-byte load per fragment.
// ---------------------------------------------------------------------------
__device__ __forceinline__ void split_store(float x, _Float16* __restrict__ hi,
                                            _Float16* __restrict__ lo, int o) {
  const _Float16 h = (_Float16)x;
  hi[o] = h;
  lo[o] = (_Float16)(x - (float)h);
}

__global__ void __launch_bounds__(256) prep_pack(
    const float* __restrict__ Wl_ab, const float* __restrict__ Wl_cb,
    const float* __restrict__ Wr_ab, const float* __restrict__ Wr_cb,
    const float* __restrict__ Wl_ac, const float* __restrict__ Wr_ac,
    _Float16* __restrict__ PabH, _Float16* __restrict__ PabL,
    _Float16* __restrict__ PcbH, _Float16* __restrict__ PcbL,
    _Float16* __restrict__ PsmH, _Float16* __restrict__ PsmL,
    _Float16* __restrict__ PacH, _Float16* __restrict__ PacL,
    _Float16* __restrict__ PrcH, _Float16* __restrict__ PrcL,
    const float* __restrict__ bl_ab, const float* __restrict__ bl_cb,
    float* __restrict__ biasB) {
  const int i = blockIdx.x * blockDim.x + threadIdx.x;
  if (i < DFEAT * DFEAT) {
    const int e   = i & 15;
    const int col = (i >> 4) & 127;
    const int hk  = i >> 11;            // kb*2 + h, 0..7
    const int h   = hk & 1;
    const int kb  = hk >> 1;
    const int k   = kb * 32 + 16 * h + e;
    const int s   = k * DFEAT + col;
    split_store(Wl_ab[s], PabH, PabL, i);
    split_store(Wl_cb[s], PcbH, PcbL, i);
    split_store(Wr_ab[s] + Wr_cb[s], PsmH, PsmL, i);
    split_store(Wl_ac[s], PacH, PacL, i);
    split_store(Wr_ac[s], PrcH, PrcL, i);
  }
  if (i < DFEAT) biasB[i] = bl_ab[i] + bl_cb[i];
}

// ---------------------------------------------------------------------------
// Conversion phase helper: thread loads 8 f32 A elements (row, cols [c0,c0+8))
// and stores f16 hi/lo runs into LDS in A-fragment order.
// A fragment (16-bit A 16x32): M = lane&15; elem e -> K = kb*32+16h+e
// (contiguous k, as in the packed B). LDS entry: plane (kb*2+h), offset m*24+sec*8.
// ---------------------------------------------------------------------------
__device__ __forceinline__ void conv_term(_Float16* __restrict__ lds,
                                          const float* __restrict__ S,
                                          int row, int c0, int dst) {
  const float* __restrict__ p = S + (size_t)row * DFEAT + c0;
  float x[8];
  *reinterpret_cast<float4*>(x)     = *reinterpret_cast<const float4*>(p);
  *reinterpret_cast<float4*>(x + 4) = *reinterpret_cast<const float4*>(p + 4);
  v8h hi, lo;
#pragma unroll
  for (int e = 0; e < 8; ++e) {
    const _Float16 h = (_Float16)x[e];
    hi[e] = h;
    lo[e] = (_Float16)(x[e] - (float)h);
  }
  *reinterpret_cast<v8h*>(lds + dst)            = hi;   // ds_store_b128
  *reinterpret_cast<v8h*>(lds + 8 * PSTR + dst) = lo;   // ds_store_b128
}

// ---------------------------------------------------------------------------
// Compute phase helper: one unscaled term S@W (K=128) from LDS A-fragments
// and packed global B-fragments; 3 f16 WMMAs per 32-wide K block.
// ---------------------------------------------------------------------------
__device__ __forceinline__ v8f compute_term(const _Float16* __restrict__ lds,
                                            const _Float16* __restrict__ BH,
                                            const _Float16* __restrict__ BL,
                                            int half, int m, int col) {
  v8f c = {0.f, 0.f, 0.f, 0.f, 0.f, 0.f, 0.f, 0.f};
#pragma unroll
  for (int kb = 0; kb < DFEAT / 32; ++kb) {
    const _Float16* __restrict__ pa = lds + (kb * 2 + half) * PSTR + m * 24;
    const v8h h0 = *reinterpret_cast<const v8h*>(pa);                 // ds_load_b128
    const v8h h1 = *reinterpret_cast<const v8h*>(pa + 8);             // ds_load_b128
    const v8h l0 = *reinterpret_cast<const v8h*>(pa + 8 * PSTR);      // ds_load_b128
    const v8h l1 = *reinterpret_cast<const v8h*>(pa + 8 * PSTR + 8);  // ds_load_b128
    const v16h ahi = __builtin_shufflevector(h0, h1, 0, 1, 2, 3, 4, 5, 6, 7,
                                             8, 9, 10, 11, 12, 13, 14, 15);
    const v16h alo = __builtin_shufflevector(l0, l1, 0, 1, 2, 3, 4, 5, 6, 7,
                                             8, 9, 10, 11, 12, 13, 14, 15);
    const size_t bo = ((size_t)(kb * 2 + half) * DFEAT + col) * 16;
    const v16h bhi = *reinterpret_cast<const v16h*>(BH + bo);  // 2x global b128
    const v16h blo = *reinterpret_cast<const v16h*>(BL + bo);  // 2x global b128
    c = __builtin_amdgcn_wmma_f32_16x16x32_f16(false, ahi, false, bhi, (short)0, c, false, false);
    c = __builtin_amdgcn_wmma_f32_16x16x32_f16(false, ahi, false, blo, (short)0, c, false, false);
    c = __builtin_amdgcn_wmma_f32_16x16x32_f16(false, alo, false, bhi, (short)0, c, false, false);
  }
  return c;
}

// ---------------------------------------------------------------------------
// Fused SAGE GEMM. 256-thread block = 8 waves; block owns 16 rows, wave w owns
// cols [16w,16w+16). Per-row mean scales (1/max(cnt,1)) applied in the
// epilogue (C/D layout: vgpr v -> M = v + 8*half, N = lane&15).
// Grids divide exactly (100000/16, 50000/16) -> EXEC all-ones for every WMMA.
// ---------------------------------------------------------------------------
template <int NT>
__global__ void __launch_bounds__(256) sage_gemm(
    const float* __restrict__ S0, const float* __restrict__ C0,
    const _Float16* __restrict__ B0H, const _Float16* __restrict__ B0L,
    const float* __restrict__ S1, const float* __restrict__ C1,
    const _Float16* __restrict__ B1H, const _Float16* __restrict__ B1L,
    const float* __restrict__ S2, const float* __restrict__ C2,
    const _Float16* __restrict__ B2H, const _Float16* __restrict__ B2L,
    const float* __restrict__ bias, float outScale, float* __restrict__ out) {
  __shared__ __align__(16) _Float16 AF[NT * 2 * 8 * PSTR];  // [term][hi/lo][plane]

  const int tid     = threadIdx.x;
  const int rowBase = blockIdx.x * 16;

  // Phase 1: block-cooperative f16 hi/lo split of the 16 A-rows of each term.
  {
    const int m   = tid >> 4;       // row within block
    const int run = tid & 15;       // which 8-col run
    const int kb  = run >> 2;
    const int r8  = run & 3;
    const int h   = r8 & 1;
    const int sec = r8 >> 1;
    const int c0  = kb * 32 + h * 16 + sec * 8;
    const int dst = (kb * 2 + h) * PSTR + m * 24 + sec * 8;
    conv_term(AF, S0, rowBase + m, c0, dst);
    if (NT > 1) conv_term(AF + 1 * 2 * 8 * PSTR, S1, rowBase + m, c0, dst);
    if (NT > 2) conv_term(AF + 2 * 2 * 8 * PSTR, S2, rowBase + m, c0, dst);
  }
  __syncthreads();

  // Phase 2: WMMA.
  const int lane = tid & 31;
  const int wave = tid >> 5;
  const int half = lane >> 4;
  const int mn   = lane & 15;
  const int col  = wave * 16 + mn;

  v8f t0 = compute_term(AF, B0H, B0L, half, mn, col);
  v8f t1 = {0.f, 0.f, 0.f, 0.f, 0.f, 0.f, 0.f, 0.f};
  v8f t2 = {0.f, 0.f, 0.f, 0.f, 0.f, 0.f, 0.f, 0.f};
  if (NT > 1) t1 = compute_term(AF + 1 * 2 * 8 * PSTR, B1H, B1L, half, mn, col);
  if (NT > 2) t2 = compute_term(AF + 2 * 2 * 8 * PSTR, B2H, B2L, half, mn, col);

  const float bv = bias[col];
#pragma unroll
  for (int v = 0; v < 8; ++v) {
    const int row = rowBase + half * 8 + v;
    float sum = t0[v] * (C0 ? 1.0f / fmaxf(C0[row], 1.0f) : 1.0f);
    if (NT > 1) sum += t1[v] * (C1 ? 1.0f / fmaxf(C1[row], 1.0f) : 1.0f);
    if (NT > 2) sum += t2[v] * (C2 ? 1.0f / fmaxf(C2[row], 1.0f) : 1.0f);
    out[(size_t)row * DFEAT + col] = (sum + bv) * outScale;
  }
}

// ---------------------------------------------------------------------------
extern "C" void kernel_launch(void* const* d_in, const int* in_sizes, int n_in,
                              void* d_out, int out_size, void* d_ws, size_t ws_size,
                              hipStream_t stream) {
  const float* x_a   = (const float*)d_in[0];
  const float* x_b   = (const float*)d_in[1];
  const float* x_c   = (const float*)d_in[2];
  const int* src_ab  = (const int*)d_in[3];
  const int* dst_ab  = (const int*)d_in[4];
  const int* src_ac  = (const int*)d_in[5];
  const int* dst_ac  = (const int*)d_in[6];
  const int* src_cb  = (const int*)d_in[7];
  const int* dst_cb  = (const int*)d_in[8];
  const float* Wl_ab = (const float*)d_in[9];
  const float* Wr_ab = (const float*)d_in[10];
  const float* Wl_ac = (const float*)d_in[11];
  const float* Wr_ac = (const float*)d_in[12];
  const float* Wl_cb = (const float*)d_in[13];
  const float* Wr_cb = (const float*)d_in[14];
  const float* bl_ab = (const float*)d_in[15];
  const float* bl_ac = (const float*)d_in[16];
  const float* bl_cb = (const float*)d_in[17];

  const int NB   = in_sizes[1] / DFEAT;  // 100000 (divisible by 16)
  const int NC   = in_sizes[2] / DFEAT;  // 50000  (divisible by 16)
  const int E_AB = in_sizes[3];
  const int E_AC = in_sizes[5];
  const int E_CB = in_sizes[7];
  const int WSZ  = DFEAT * DFEAT;        // 16384 elements per weight plane

  // Workspace layout: f32 acc/cnt region (zeroed per call), then 10 f16 weight
  // planes (hi/lo for Wl_ab, Wl_cb, Wr_ab+Wr_cb, Wl_ac, Wr_ac), then bias_b.
  float* ws     = (float*)d_ws;
  float* acc_ab = ws;
  float* acc_cb = acc_ab + (size_t)NB * DFEAT;
  float* acc_ac = acc_cb + (size_t)NB * DFEAT;
  float* cnt_ab = acc_ac + (size_t)NC * DFEAT;
  float* cnt_cb = cnt_ab + NB;
  float* cnt_ac = cnt_cb + NB;
  _Float16* PabH = (_Float16*)(cnt_ac + NC);
  _Float16* PabL = PabH + WSZ;
  _Float16* PcbH = PabL + WSZ;
  _Float16* PcbL = PcbH + WSZ;
  _Float16* PsmH = PcbL + WSZ;
  _Float16* PsmL = PsmH + WSZ;
  _Float16* PacH = PsmL + WSZ;
  _Float16* PacL = PacH + WSZ;
  _Float16* PrcH = PacL + WSZ;
  _Float16* PrcL = PrcH + WSZ;
  float* bias_b  = (float*)(PrcL + WSZ);

  // Zero accumulators + counts every call (harness poisons ws once only).
  const size_t zeroFloats =
      (size_t)NB * DFEAT * 2 + (size_t)NC * DFEAT + (size_t)NB * 2 + (size_t)NC;
  hipMemsetAsync(ws, 0, zeroFloats * sizeof(float), stream);

  prep_pack<<<(WSZ + 255) / 256, 256, 0, stream>>>(
      Wl_ab, Wl_cb, Wr_ab, Wr_cb, Wl_ac, Wr_ac,
      PabH, PabL, PcbH, PcbL, PsmH, PsmL, PacH, PacL, PrcH, PrcL,
      bl_ab, bl_cb, bias_b);

  agg_edges<<<(E_AB + 7) / 8, 256, 0, stream>>>(x_a, src_ab, dst_ab, acc_ab, cnt_ab, E_AB);
  agg_edges<<<(E_CB + 7) / 8, 256, 0, stream>>>(x_c, src_cb, dst_cb, acc_cb, cnt_cb, E_CB);
  agg_edges<<<(E_AC + 7) / 8, 256, 0, stream>>>(x_a, src_ac, dst_ac, acc_ac, cnt_ac, E_AC);

  float* z_b = (float*)d_out;             // [NB,128]
  float* z_c = z_b + (size_t)NB * DFEAT;  // [NC,128]

  // z_b: 3 fused terms (two relation means + shared x_b term), 0.5 relation-mean.
  sage_gemm<3><<<NB / 16, 256, 0, stream>>>(acc_ab, cnt_ab, PabH, PabL,
                                            acc_cb, cnt_cb, PcbH, PcbL,
                                            x_b, nullptr, PsmH, PsmL,
                                            bias_b, 0.5f, z_b);
  // z_c: 2 terms.
  sage_gemm<2><<<NC / 16, 256, 0, stream>>>(acc_ac, cnt_ac, PacH, PacL,
                                            x_c, nullptr, PrcH, PrcL,
                                            nullptr, nullptr, nullptr, nullptr,
                                            bl_ac, 1.0f, z_c);
}
